// GNNModel_52639119179830
// MI455X (gfx1250) — matrix operand order repaired and checked by reading, not compiled, not run on hardware
//
#include <hip/hip_runtime.h>
#include <math.h>
#include <cstdint>

// ---------------------------------------------------------------------------
// Types for CDNA5 WMMA (wave32) + TDM descriptors
// ---------------------------------------------------------------------------
typedef __attribute__((ext_vector_type(16))) __bf16 v16bf;
typedef __attribute__((ext_vector_type(8)))  float  v8f;
typedef __attribute__((ext_vector_type(4)))  unsigned int u32x4;
typedef __attribute__((ext_vector_type(4)))  int    i32x4;
typedef __attribute__((ext_vector_type(8)))  int    i32x8;

#define DIN 192
#define HF  64

// ---------------------------------------------------------------------------
// Utility kernels
// ---------------------------------------------------------------------------
__global__ void fill_f32(float* __restrict__ p, float v, long n) {
    long i = (long)blockIdx.x * blockDim.x + threadIdx.x;
    if (i < n) p[i] = v;
}

__global__ void edge_deg_kernel(const int* __restrict__ src,
                                const int* __restrict__ dst,
                                float* __restrict__ deg, int E) {
    int e = blockIdx.x * blockDim.x + threadIdx.x;
    if (e >= E) return;
    atomicAdd(&deg[src[e]], 1.0f);
    atomicAdd(&deg[dst[e]], 1.0f);
}

__global__ void rsqrt_kernel(float* __restrict__ d, int n) {
    int i = blockIdx.x * blockDim.x + threadIdx.x;
    if (i < n) d[i] = rsqrtf(d[i]);
}

// Convert W[K,64] f32 (row-major) -> Wt[64,K] bf16 (transposed) in global mem.
__global__ void transpose_w_bf16(const float* __restrict__ W,
                                 __bf16* __restrict__ Wt, int K) {
    int i = blockIdx.x * blockDim.x + threadIdx.x;  // over K*64
    if (i >= K * 64) return;
    int k = i >> 6;
    int c = i & 63;
    Wt[c * K + k] = (__bf16)W[i];
}

// ---------------------------------------------------------------------------
// WMMA GEMM: Y[N,64] = act(X[N,K] @ W[K,64] (+ b))
// Block = 256 threads = 8 waves; each wave owns a 16-row tile x 64 cols.
// Weights (pre-transposed bf16 in global) staged into LDS via TDM
// tensor_load_to_lds (wave 0 issues, s_wait_tensorcnt, barrier).
// ---------------------------------------------------------------------------
template <int K, bool RELU_BIAS>
__global__ void gemm_node_kernel(const float* __restrict__ X,
                                 const __bf16* __restrict__ Wt,  // [64,K] bf16
                                 const float* __restrict__ bias, // [64] or null
                                 float* __restrict__ Y,          // [N,64]
                                 int nrows) {
    __shared__ __bf16 Wlds[K * 64];   // transposed: Wlds[c*K + k]

    if (threadIdx.x < 32) {
        // --- TDM: single 1D descriptor copy of 64*K bf16 elements to LDS ---
        const unsigned long long ga = (unsigned long long)(uintptr_t)Wt;
        const unsigned int lds_off  = (unsigned int)(uintptr_t)&Wlds[0];
        const unsigned int nelem    = (unsigned int)(K * 64); // <= 12288
        u32x4 g0;
        g0.x = 1u;                                    // count=1, user descriptor
        g0.y = lds_off;                               // lds_addr (bytes)
        g0.z = (unsigned int)(ga & 0xFFFFFFFFu);      // global_addr[31:0]
        g0.w = (unsigned int)((ga >> 32) & 0x01FFFFFFu) | (2u << 30); // [56:32]|type=2
        i32x8 g1;
        g1[0] = 0x00010000;                           // data_size=1 -> 2-byte elems
        g1[1] = (int)((nelem & 0xFFFFu) << 16);       // tensor_dim0[15:0]
        g1[2] = (int)(((nelem >> 16) & 0xFFFFu) | (1u << 16)); // td0[31:16], td1=1
        g1[3] = (int)((nelem & 0xFFFFu) << 16);       // tile_dim0 = nelem
        g1[4] = 0;                                    // tile_dim1/2 unused
        g1[5] = (int)nelem;                           // tensor_dim0_stride
        g1[6] = 0;
        g1[7] = 0;
        i32x4 z4 = {0, 0, 0, 0};
#if __clang_major__ >= 23
        i32x8 z8 = {0, 0, 0, 0, 0, 0, 0, 0};
        __builtin_amdgcn_tensor_load_to_lds(g0, g1, z4, z4, z8, 0);
#else
        __builtin_amdgcn_tensor_load_to_lds(g0, g1, z4, z4, 0);
#endif
        __builtin_amdgcn_s_wait_tensorcnt(0);
    }
    __syncthreads();

    const int wave = threadIdx.x >> 5;
    const int lane = threadIdx.x & 31;
    const int row0 = (blockIdx.x * 8 + wave) * 16;
    if (row0 >= nrows) return;            // wave-uniform: EXEC stays all-ones

    const int r  = lane & 15;             // A: row within tile; B/D: column
    const int hi = lane >> 4;             // lane group (0/1)
    int arow = row0 + r;
    if (arow >= nrows) arow = nrows - 1;  // clamp (partial tiles guarded below)
    const float* xrow = X + (size_t)arow * K;

    v8f acc[4] = {};                      // 4 N-tiles of 16 cols

    for (int kt = 0; kt < K / 32; ++kt) {
        const int kbase = kt * 32 + hi * 16;
        if (kt + 1 < K / 32)              // prefetch next A k-tile
            __builtin_prefetch(xrow + kbase + 32, 0, 1);
        // ---- A fragment (16x32 bf16): two aligned float4 pairs per lane ----
        const float4 f0 = *(const float4*)(xrow + kbase);
        const float4 f1 = *(const float4*)(xrow + kbase + 4);
        const float4 f2 = *(const float4*)(xrow + kbase + 8);
        const float4 f3 = *(const float4*)(xrow + kbase + 12);
        v16bf a;
        a[0]  = (__bf16)f0.x; a[1]  = (__bf16)f0.y; a[2]  = (__bf16)f0.z; a[3]  = (__bf16)f0.w;
        a[4]  = (__bf16)f1.x; a[5]  = (__bf16)f1.y; a[6]  = (__bf16)f1.z; a[7]  = (__bf16)f1.w;
        a[8]  = (__bf16)f2.x; a[9]  = (__bf16)f2.y; a[10] = (__bf16)f2.z; a[11] = (__bf16)f2.w;
        a[12] = (__bf16)f3.x; a[13] = (__bf16)f3.y; a[14] = (__bf16)f3.z; a[15] = (__bf16)f3.w;

        // ---- B fragments (32x16 bf16) from LDS, 4 column tiles ----
        const int kb = kt * 32 + hi * 16;
#pragma unroll
        for (int nt = 0; nt < 4; ++nt) {
            const __bf16* wcol = &Wlds[(size_t)(nt * 16 + r) * K + kb];
            v16bf bfr;
#pragma unroll
            for (int v = 0; v < 16; ++v) bfr[v] = wcol[v];
            acc[nt] = __builtin_amdgcn_wmma_f32_16x16x32_bf16(
                false, a, false, bfr, (short)0, acc[nt], false, false);
        }
    }

    // ---- Epilogue: lane -> column (r), VGPR v -> row v + 8*hi ----
    float bv[4];
    if constexpr (RELU_BIAS) {
#pragma unroll
        for (int nt = 0; nt < 4; ++nt) bv[nt] = bias[nt * 16 + r];
    }

    if (row0 + 16 <= nrows) {             // full tile: no per-element guards
#pragma unroll
        for (int nt = 0; nt < 4; ++nt) {
            float* ycol = Y + ((size_t)(row0 + 8 * hi)) * 64 + nt * 16 + r;
#pragma unroll
            for (int v = 0; v < 8; ++v) {
                float val = acc[nt][v];
                if constexpr (RELU_BIAS) val = fmaxf(val + bv[nt], 0.0f);
                ycol[(size_t)v * 64] = val;
            }
        }
    } else {                              // ragged tail (unused when N%16==0)
#pragma unroll
        for (int nt = 0; nt < 4; ++nt) {
            const int col = nt * 16 + r;
#pragma unroll
            for (int v = 0; v < 8; ++v) {
                const int orow = row0 + v + 8 * hi;
                if (orow < nrows) {
                    float val = acc[nt][v];
                    if constexpr (RELU_BIAS) val = fmaxf(val + bv[nt], 0.0f);
                    Y[(size_t)orow * 64 + col] = val;
                }
            }
        }
    }
}

// ---------------------------------------------------------------------------
// Edge scatter: agg[d] += hw[s]*c ; agg[s] += hw[d]*c ; c = dis[s]*dis[d]
// ---------------------------------------------------------------------------
__global__ void scatter_kernel(const int* __restrict__ src,
                               const int* __restrict__ dst,
                               const float* __restrict__ dis,
                               const float* __restrict__ hw,
                               float* __restrict__ agg, int E) {
    long tid = (long)blockIdx.x * blockDim.x + threadIdx.x;
    int e = (int)(tid >> 4);
    int q = (int)(tid & 15);
    if (e >= E) return;
    const int s = src[e], d = dst[e];
    const float c = dis[s] * dis[d];
    const float4 vs = *(const float4*)(hw + (size_t)s * 64 + q * 4);
    const float4 vd = *(const float4*)(hw + (size_t)d * 64 + q * 4);
    float* ad = agg + (size_t)d * 64 + q * 4;
    float* as = agg + (size_t)s * 64 + q * 4;
    atomicAdd(ad + 0, vs.x * c); atomicAdd(ad + 1, vs.y * c);
    atomicAdd(ad + 2, vs.z * c); atomicAdd(ad + 3, vs.w * c);
    atomicAdd(as + 0, vd.x * c); atomicAdd(as + 1, vd.y * c);
    atomicAdd(as + 2, vd.z * c); atomicAdd(as + 3, vd.w * c);
}

// h = relu(agg + hw * dis^2 + b)
__global__ void conv_finish_kernel(const float* __restrict__ agg,
                                   const float* __restrict__ hw,
                                   const float* __restrict__ dis,
                                   const float* __restrict__ bias,
                                   float* __restrict__ h, int n) {
    long tid = (long)blockIdx.x * blockDim.x + threadIdx.x;
    int node = (int)(tid >> 4);
    int q = (int)(tid & 15);
    if (node >= n) return;
    const float di = dis[node];
    const float d2 = di * di;
    const float4 a4 = *(const float4*)(agg + (size_t)node * 64 + q * 4);
    const float4 w4 = *(const float4*)(hw + (size_t)node * 64 + q * 4);
    const float4 b4 = *(const float4*)(bias + q * 4);
    float4 r;
    r.x = fmaxf(a4.x + w4.x * d2 + b4.x, 0.0f);
    r.y = fmaxf(a4.y + w4.y * d2 + b4.y, 0.0f);
    r.z = fmaxf(a4.z + w4.z * d2 + b4.z, 0.0f);
    r.w = fmaxf(a4.w + w4.w * d2 + b4.w, 0.0f);
    *(float4*)(h + (size_t)node * 64 + q * 4) = r;
}

// ---------------------------------------------------------------------------
// Mean pooling over batch ids
// ---------------------------------------------------------------------------
__global__ void pool_sum_kernel(const float* __restrict__ h,
                                const int* __restrict__ batch,
                                float* __restrict__ sums, int n) {
    long tid = (long)blockIdx.x * blockDim.x + threadIdx.x;
    int node = (int)(tid >> 4);
    int q = (int)(tid & 15);
    if (node >= n) return;
    const int g = batch[node];
    const float4 v = *(const float4*)(h + (size_t)node * 64 + q * 4);
    float* p = sums + (size_t)g * 64 + q * 4;
    atomicAdd(p + 0, v.x); atomicAdd(p + 1, v.y);
    atomicAdd(p + 2, v.z); atomicAdd(p + 3, v.w);
}

__global__ void pool_cnt_kernel(const int* __restrict__ batch,
                                float* __restrict__ cnt, int n) {
    int i = blockIdx.x * blockDim.x + threadIdx.x;
    if (i < n) atomicAdd(&cnt[batch[i]], 1.0f);
}

// ---------------------------------------------------------------------------
// Head: out = log_softmax(relu(pooled @ Wf1 + bf1) @ Wf2 + bf2)
// ---------------------------------------------------------------------------
__global__ void head_kernel(const float* __restrict__ sums,
                            const float* __restrict__ cnt,
                            const float* __restrict__ Wf1,  // [64,32]
                            const float* __restrict__ bf1,  // [32]
                            const float* __restrict__ Wf2,  // [32,2]
                            const float* __restrict__ bf2,  // [2]
                            float* __restrict__ out) {      // [G,2]
    const int g = blockIdx.x;
    const int t = threadIdx.x;
    __shared__ float p[64];
    __shared__ float o1[32];
    __shared__ float lg[2];

    const float c = fmaxf(cnt[g], 1.0f);
    p[t] = sums[(size_t)g * 64 + t] / c;
    __syncthreads();

    if (t < 32) {
        float acc = bf1[t];
#pragma unroll
        for (int k = 0; k < 64; ++k) acc += p[k] * Wf1[k * 32 + t];
        o1[t] = fmaxf(acc, 0.0f);
    }
    __syncthreads();

    if (t < 2) {
        float acc = bf2[t];
#pragma unroll
        for (int k = 0; k < 32; ++k) acc += o1[k] * Wf2[k * 2 + t];
        lg[t] = acc;
    }
    __syncthreads();

    if (t < 2) {
        const float m = fmaxf(lg[0], lg[1]);
        const float lse = m + logf(expf(lg[0] - m) + expf(lg[1] - m));
        out[g * 2 + t] = lg[t] - lse;
    }
}

// ---------------------------------------------------------------------------
// Host launch
// ---------------------------------------------------------------------------
static inline size_t alignup(size_t x) { return (x + 255) & ~(size_t)255; }

extern "C" void kernel_launch(void* const* d_in, const int* in_sizes, int n_in,
                              void* d_out, int out_size, void* d_ws, size_t ws_size,
                              hipStream_t stream) {
    const int N = in_sizes[0] / DIN;
    const int E = in_sizes[1] / 2;
    const int G = out_size / 2;

    const float* x     = (const float*)d_in[0];
    const int*   src   = (const int*)d_in[1];
    const int*   dst   = src + E;
    const int*   batch = (const int*)d_in[2];
    const float* W_in  = (const float*)d_in[4];
    const float* b_in  = (const float*)d_in[5];
    const float* W1    = (const float*)d_in[6];
    const float* b1    = (const float*)d_in[7];
    const float* W2    = (const float*)d_in[8];
    const float* b2    = (const float*)d_in[9];
    const float* Wf1   = (const float*)d_in[10];
    const float* bf1   = (const float*)d_in[11];
    const float* Wf2   = (const float*)d_in[12];
    const float* bf2   = (const float*)d_in[13];
    float* out = (float*)d_out;

    // Workspace carve-up
    char* w = (char*)d_ws;
    float*  dis   = (float*)w;  w += alignup((size_t)N * 4);
    float*  h     = (float*)w;  w += alignup((size_t)N * 64 * 4);
    float*  hw    = (float*)w;  w += alignup((size_t)N * 64 * 4);
    float*  agg   = (float*)w;  w += alignup((size_t)N * 64 * 4);
    float*  sums  = (float*)w;  w += alignup((size_t)G * 64 * 4);
    float*  cnt   = (float*)w;  w += alignup((size_t)G * 4);
    __bf16* wt_in = (__bf16*)w; w += alignup((size_t)DIN * 64 * 2);
    __bf16* wt1   = (__bf16*)w; w += alignup((size_t)HF * 64 * 2);
    __bf16* wt2   = (__bf16*)w; w += alignup((size_t)HF * 64 * 2);
    (void)ws_size; (void)n_in;

    const int T = 256;
    const long NF = (long)N * 64;

    // 0) Weight precompute: f32 [K,64] -> bf16 transposed [64,K]
    transpose_w_bf16<<<(DIN * 64 + T - 1) / T, T, 0, stream>>>(W_in, wt_in, DIN);
    transpose_w_bf16<<<(HF * 64 + T - 1) / T, T, 0, stream>>>(W1, wt1, HF);
    transpose_w_bf16<<<(HF * 64 + T - 1) / T, T, 0, stream>>>(W2, wt2, HF);

    // 1) Degrees -> dis = rsqrt(deg)
    fill_f32<<<(N + T - 1) / T, T, 0, stream>>>(dis, 1.0f, N);
    edge_deg_kernel<<<(E + T - 1) / T, T, 0, stream>>>(src, dst, dis, E);
    rsqrt_kernel<<<(N + T - 1) / T, T, 0, stream>>>(dis, N);

    // 2) Input layer: h = relu(x @ W_in + b_in)
    gemm_node_kernel<DIN, true><<<(N + 127) / 128, T, 0, stream>>>(x, wt_in, b_in, h, N);

    // 3) GCN conv layer 1
    gemm_node_kernel<HF, false><<<(N + 127) / 128, T, 0, stream>>>(h, wt1, nullptr, hw, N);
    fill_f32<<<(unsigned)((NF + T - 1) / T), T, 0, stream>>>(agg, 0.0f, NF);
    scatter_kernel<<<(unsigned)(((long)E * 16 + T - 1) / T), T, 0, stream>>>(src, dst, dis, hw, agg, E);
    conv_finish_kernel<<<(unsigned)(((long)N * 16 + T - 1) / T), T, 0, stream>>>(agg, hw, dis, b1, h, N);

    // 4) GCN conv layer 2
    gemm_node_kernel<HF, false><<<(N + 127) / 128, T, 0, stream>>>(h, wt2, nullptr, hw, N);
    fill_f32<<<(unsigned)((NF + T - 1) / T), T, 0, stream>>>(agg, 0.0f, NF);
    scatter_kernel<<<(unsigned)(((long)E * 16 + T - 1) / T), T, 0, stream>>>(src, dst, dis, hw, agg, E);
    conv_finish_kernel<<<(unsigned)(((long)N * 16 + T - 1) / T), T, 0, stream>>>(agg, hw, dis, b2, h, N);

    // 5) Global mean pool
    fill_f32<<<(G * 64 + T - 1) / T, T, 0, stream>>>(sums, 0.0f, (long)G * 64);
    fill_f32<<<(G + T - 1) / T, T, 0, stream>>>(cnt, 0.0f, G);
    pool_sum_kernel<<<(unsigned)(((long)N * 16 + T - 1) / T), T, 0, stream>>>(h, batch, sums, N);
    pool_cnt_kernel<<<(N + T - 1) / T, T, 0, stream>>>(batch, cnt, N);

    // 6) MLP head + log_softmax
    head_kernel<<<G, 64, 0, stream>>>(sums, cnt, Wf1, bf1, Wf2, bf2, out);
}
